// SRUCell_39359080300755
// MI455X (gfx1250) — matrix operand order repaired and checked by reading, not compiled, not run on hardware
//
#include <hip/hip_runtime.h>

// ---------------------------------------------------------------------------
// SRU on MI455X (gfx1250).
// Phase 1: bf16 WMMA GEMM u = x@W (M=16384,K=2048,N=6144) with fused
//          bias+sigmoid; writes candidate/gates as three planar fp16 arrays
//          (192 MB in d_ws) with non-temporal stores so x/W stay L2-resident.
//          All global accesses use uniform SGPR base + 32-bit VGPR offsets so
//          they lower to global_* (not flat_*) and DScnt stays LDS-only.
// Phase 2: per-(b,d) recurrence over L (fp32 state), coalesced NT fp16 reads.
// ---------------------------------------------------------------------------

typedef __bf16 bf16_t;
typedef bf16_t v16bf __attribute__((ext_vector_type(16)));
typedef float  v8f   __attribute__((ext_vector_type(8)));
typedef unsigned short v8u16  __attribute__((ext_vector_type(8)));
typedef unsigned short v16u16 __attribute__((ext_vector_type(16)));
typedef _Float16 f16;

static constexpr int Lseq = 1024;
static constexpr int Bsz  = 16;
static constexpr int Din  = 2048;
static constexpr int Dd   = 2048;
static constexpr int M    = Lseq * Bsz;   // 16384
static constexpr int N    = 3 * Dd;       // 6144
static constexpr int Kdim = Din;          // 2048
static constexpr unsigned PL = (unsigned)M * (unsigned)Dd;  // fp16 plane elems

static constexpr int BM  = 128;           // block tile M
static constexpr int BN  = 128;           // block tile N
static constexpr int KT  = 32;            // k-step == WMMA K
static constexpr int LDP = KT + 8;        // LDS row pitch (bf16 elems), 80 B

// round-half-up fp32->bf16 (1 VALU/elem); pack two into a dword with v_perm
__device__ __forceinline__ unsigned pack_bf16x2(float lo, float hi) {
  unsigned a = __builtin_bit_cast(unsigned, lo) + 0x8000u;
  unsigned b = __builtin_bit_cast(unsigned, hi) + 0x8000u;
  return __builtin_amdgcn_perm(b, a, 0x07060302u);  // {b.hi16, a.hi16}
}
__device__ __forceinline__ unsigned short bf16h(float f) {
  return (unsigned short)((__builtin_bit_cast(unsigned, f) + 0x8000u) >> 16);
}

__device__ __forceinline__ v16bf frag16(const unsigned short* p0,
                                        const unsigned short* p1) {
  v8u16 lo = *(const v8u16*)p0;   // ds_load_b128
  v8u16 hi = *(const v8u16*)p1;   // ds_load_b128
  v16u16 t;
#pragma unroll
  for (int i = 0; i < 8; ++i) { t[i] = lo[i]; t[i + 8] = hi[i]; }
  return __builtin_bit_cast(v16bf, t);
}

__device__ __forceinline__ float sigmoidf_fast(float v) {
  return 1.0f / (1.0f + __expf(-v));
}

__global__ void __launch_bounds__(256)
sru_gemm_gates(const float* __restrict__ X,     // [M, Kdim]
               const float* __restrict__ W,     // [Kdim, N]
               const float* __restrict__ bias,  // [2*Dd]
               f16* __restrict__ Uws)           // 3 planes of [M, Dd]
{
  __shared__ __align__(16) unsigned short As[BM][LDP];  // [m][k]
  __shared__ __align__(16) unsigned short Bs[BN][LDP];  // [n][k] (transposed)

  const int tid   = threadIdx.x;
  const int lane  = tid & 31;
  const int wave  = tid >> 5;       // 0..7
  const int waveM = wave >> 1;      // 0..3 -> 32 rows each
  const int waveN = wave & 1;       // 0..1 -> 64 cols each
  const int l     = lane & 15;
  const int hi    = lane >> 4;      // half-wave select

  const int blockN = blockIdx.x * BN;
  const int blockM = blockIdx.y * BM;

  // uniform bases (stay in SGPRs) + per-thread 32-bit element offsets
  const float* aBase = X + (size_t)blockM * Kdim;
  const float* bBase = W + blockN;

  unsigned aOff[4], bOff[4];
  unsigned short* aSt[4];
  unsigned short* bSt[4];
#pragma unroll
  for (int i = 0; i < 4; ++i) {
    int idx = i * 256 + tid;                 // 1024 float4 slots
    int m = idx >> 3, kq = idx & 7;          // A: 128 rows x 8 quads
    aOff[i] = (unsigned)m * Kdim + kq * 4;
    aSt[i]  = &As[m][kq * 4];
    int kk = idx >> 5, nq = idx & 31;        // B: 32 k-rows x 32 quads
    bOff[i] = (unsigned)kk * N + nq * 4;
    bSt[i]  = &Bs[nq * 4][kk];
  }

  v8f acc[2][4] = {};
  float4 aReg[4], bReg[4];

#pragma unroll
  for (int i = 0; i < 4; ++i) {
    aReg[i] = *(const float4*)(aBase + aOff[i]);  aOff[i] += KT;
    bReg[i] = *(const float4*)(bBase + bOff[i]);  bOff[i] += (unsigned)KT * N;
  }

  const int nkt = Kdim / KT;  // 64
  for (int kt = 0; kt < nkt; ++kt) {
    __syncthreads();
    // regs -> LDS, fp32 -> bf16
#pragma unroll
    for (int i = 0; i < 4; ++i) {
      uint2 p;
      p.x = pack_bf16x2(aReg[i].x, aReg[i].y);
      p.y = pack_bf16x2(aReg[i].z, aReg[i].w);
      *(uint2*)aSt[i] = p;                              // ds_store_b64
    }
#pragma unroll
    for (int i = 0; i < 4; ++i) {                       // transposed scatter
      bSt[i][0 * LDP] = bf16h(bReg[i].x);
      bSt[i][1 * LDP] = bf16h(bReg[i].y);
      bSt[i][2 * LDP] = bf16h(bReg[i].z);
      bSt[i][3 * LDP] = bf16h(bReg[i].w);
    }
    __syncthreads();

    if (kt + 1 < nkt) {                                 // overlap with WMMA
#pragma unroll
      for (int i = 0; i < 4; ++i) {
        aReg[i] = *(const float4*)(aBase + aOff[i]);  aOff[i] += KT;
        bReg[i] = *(const float4*)(bBase + bOff[i]);  bOff[i] += (unsigned)KT * N;
      }
    }

    // fragments per ISA 16-bit A(16x32)/B(32x16) lane maps
    v16bf afrag[2], bfrag[4];
#pragma unroll
    for (int mt = 0; mt < 2; ++mt) {
      const unsigned short* row = &As[waveM * 32 + mt * 16 + l][0];
      afrag[mt] = frag16(row + hi * 8, row + 16 + hi * 8);
    }
#pragma unroll
    for (int nt = 0; nt < 4; ++nt) {
      const unsigned short* col = &Bs[waveN * 64 + nt * 16 + l][0];
      bfrag[nt] = frag16(col + hi * 8, col + 16 + hi * 8);
    }
#pragma unroll
    for (int mt = 0; mt < 2; ++mt)
#pragma unroll
      for (int nt = 0; nt < 4; ++nt)
        acc[mt][nt] = __builtin_amdgcn_wmma_f32_16x16x32_bf16(
            false, afrag[mt], false, bfrag[nt],
            (short)0, acc[mt][nt], false, false);
  }

  // ---- epilogue: bias + sigmoid on gate columns ----
  // Single uniform base Uws; plane chosen by 32-bit offset kk*PL so stores
  // stay global_store_d16 (saddr form) with NT hint.
#pragma unroll
  for (int nt = 0; nt < 4; ++nt) {
    const int nn = blockN + waveN * 64 + nt * 16 + l;   // global u-column
    const int kk = nn % 3;
    const int dd = nn / 3;
    float badd = 0.0f;
    bool gate = false;
    if (kk == 1)      { badd = bias[dd];      gate = true; }
    else if (kk == 2) { badd = bias[Dd + dd]; gate = true; }
    const unsigned plOff = (unsigned)kk * PL + (unsigned)dd;
#pragma unroll
    for (int mt = 0; mt < 2; ++mt) {
#pragma unroll
      for (int r = 0; r < 8; ++r) {
        const int m = blockM + waveM * 32 + mt * 16 + hi * 8 + r;
        float v = acc[mt][nt][r];
        if (gate) v = sigmoidf_fast(v + badd);
        __builtin_nontemporal_store((f16)v, Uws + (plOff + (unsigned)m * Dd));
      }
    }
  }
}

__global__ void __launch_bounds__(256)
sru_scan(const f16* __restrict__ Uws,  // 3 planes of [L*B, Dd]
         const float* __restrict__ X,  // [L, B, Din]
         const float* __restrict__ C0, // [B, Dd]
         float* __restrict__ H,        // [L, B, Dd]
         float* __restrict__ Clast)    // [B, Dd]
{
  const int idx = blockIdx.x * blockDim.x + threadIdx.x;   // 0..B*Dd-1
  const int b = idx >> 11;            // /Dd (2048)
  const int d = idx & (Dd - 1);

  float c = C0[idx];
  const unsigned st = (unsigned)Bsz * Dd;    // per-timestep plane stride
  unsigned uo = (unsigned)b * Dd + d;        // offset within a plane
  unsigned xo = (unsigned)b * Din + d;       // offset within X / H

  const f16* P0 = Uws;
  const f16* P1 = Uws + PL;
  const f16* P2 = Uws + 2u * PL;

  for (int t = 0; t < Lseq; ++t) {
    const float u0 = (float)__builtin_nontemporal_load(P0 + uo);
    const float g1 = (float)__builtin_nontemporal_load(P1 + uo);
    const float g2 = (float)__builtin_nontemporal_load(P2 + uo);
    c = (c - u0) * g1 + u0;
    const float xt = X[xo];
    __builtin_nontemporal_store((tanhf(c) - xt) * g2 + xt, H + xo);
    uo += st;
    xo += (unsigned)Bsz * Din;
  }
  Clast[idx] = c;
}

extern "C" void kernel_launch(void* const* d_in, const int* in_sizes, int n_in,
                              void* d_out, int out_size, void* d_ws, size_t ws_size,
                              hipStream_t stream) {
  (void)in_sizes; (void)n_in; (void)out_size; (void)ws_size;
  const float* X    = (const float*)d_in[0];   // [L,B,Din]
  const float* C0   = (const float*)d_in[1];   // [B,Dd]
  const float* W    = (const float*)d_in[2];   // [Din, 3*Dd]
  const float* bias = (const float*)d_in[3];   // [2*Dd]

  float* H     = (float*)d_out;                       // [L,B,Dd]
  float* Clast = H + (size_t)Lseq * Bsz * Dd;         // [B,Dd]

  f16* Uws = (f16*)d_ws;                              // 3 planes, 192 MB total

  dim3 g1(N / BN, M / BM);   // (48, 128)
  sru_gemm_gates<<<g1, dim3(256), 0, stream>>>(X, W, bias, Uws);

  dim3 g2((Bsz * Dd) / 256); // 128 blocks
  sru_scan<<<g2, dim3(256), 0, stream>>>(Uws, X, C0, H, Clast);
}
